// FlashMultiHeadAttention_13872744366835
// MI455X (gfx1250) — matrix operand order repaired and checked by reading, not compiled, not run on hardware
//
#include <hip/hip_runtime.h>

// ---------------------------------------------------------------------------
// FlashMultiHeadAttention for MI455X (gfx1250, wave32, WMMA 16x16x32 f16,
// Tensor Data Mover for LDS tile staging, double-buffered GEMM).
// B=4, L=2048, D=2048, H=16, HD=128
// ---------------------------------------------------------------------------

typedef _Float16 v8h  __attribute__((ext_vector_type(8)));
typedef _Float16 v16h __attribute__((ext_vector_type(16)));
typedef float    v8f  __attribute__((ext_vector_type(8)));
typedef unsigned int u32x4 __attribute__((ext_vector_type(4)));
typedef int          i32x4 __attribute__((ext_vector_type(4)));
typedef int          i32x8 __attribute__((ext_vector_type(8)));

#define Bsz 4
#define Lsz 2048
#define Dsz 2048
#define Hsz 16
#define HDsz 128
#define SCALE 0.088388347648318440550f /* 1/sqrt(128) */

#if defined(__HIP_DEVICE_COMPILE__) && __has_builtin(__builtin_amdgcn_tensor_load_to_lds)
#define HAVE_TDM 1
#else
#define HAVE_TDM 0
#endif

#define LDS_OFF(p) ((unsigned)(unsigned long long)(const void*)(p))

#if HAVE_TDM
// 2-D TDM load: tile_w x tile_h of 2-byte elements, row stride in elements.
// D# group0: count=1 | lds_addr | global_addr[56:0] | type=2  (ISA 8.3)
// D# group1: data_size=2B, tensor_dim0/1 = tile dims, dim0_stride (ISA 8.4)
__device__ __forceinline__ void tdm_load_2d(unsigned lds_off, const void* gp,
                                            unsigned tile_w, unsigned tile_h,
                                            unsigned long long row_stride_elems) {
  unsigned long long ga = (unsigned long long)gp;
  u32x4 g0;
  g0.x = 1u;                                            // count=1, user mode
  g0.y = lds_off;                                       // lds_addr (bytes)
  g0.z = (unsigned)ga;                                  // global_addr[31:0]
  g0.w = (unsigned)((ga >> 32) & 0x01ffffffu) | 0x80000000u; // ga[56:32]|type=2
  i32x8 g1;
  g1[0] = 0x00010000;                                   // data_size=1 (2 bytes)
  g1[1] = (int)(tile_w << 16);                          // tensor_dim0[15:0]
  g1[2] = (int)((tile_w >> 16) | (tile_h << 16));       // dim0 hi | tensor_dim1 lo
  g1[3] = (int)((tile_h >> 16) | (tile_w << 16));       // dim1 hi | tile_dim0
  g1[4] = (int)tile_h;                                  // tile_dim1, tile_dim2=0
  g1[5] = (int)(unsigned)row_stride_elems;              // dim0_stride[31:0]
  g1[6] = (int)(unsigned)(row_stride_elems >> 32);      // dim0_stride[47:32]
  g1[7] = 0;                                            // dim1_stride (unused, 2D)
  i32x4 z4 = {0, 0, 0, 0};
#if defined(__clang_major__) && __clang_major__ >= 23
  i32x8 z8 = {};
  __builtin_amdgcn_tensor_load_to_lds(g0, g1, z4, z4, z8, 0);
#else
  __builtin_amdgcn_tensor_load_to_lds(g0, g1, z4, z4, 0);
#endif
}
#endif

__device__ __forceinline__ v16h cat16(v8h lo, v8h hi) {
  return __builtin_shufflevector(lo, hi, 0,1,2,3,4,5,6,7,8,9,10,11,12,13,14,15);
}

// A-fragment (16x32 f16), row-major [16+][ld] halves, base at (row0,k0).
// M=lane%16; halves 0..7 -> K=(lane<16?0:8)+h, halves 8..15 -> K=(lane<16?16:24)+h.
__device__ __forceinline__ v16h load_a_frag(const _Float16* base, int ld, int lane) {
  const int m  = lane & 15;
  const int hi = lane >> 4;
  const _Float16* p = base + m * ld;
  v8h c0 = *(const v8h*)(p + (hi ? 8 : 0));
  v8h c1 = *(const v8h*)(p + (hi ? 24 : 16));
  return cat16(c0, c1);
}

// B-fragment (32x16 f16), source stored transposed: [N][ld], contraction
// contiguous. N=lane%16; halves 0..7 -> K=(lane<16?0:8)+h, halves 8..15 -> K=(lane<16?16:24)+h.
__device__ __forceinline__ v16h load_b_frag(const _Float16* base, int ld, int lane) {
  const int n  = lane & 15;
  const int hi = lane >> 4;
  const _Float16* p = base + n * ld;
  v8h c0 = *(const v8h*)(p + (hi ? 8 : 0));
  v8h c1 = *(const v8h*)(p + (hi ? 24 : 16));
  return cat16(c0, c1);
}

// ---------------------------------------------------------------------------
// f32 -> f16 convert / transpose-convert
// ---------------------------------------------------------------------------
__global__ void convert_f32_f16(const float* __restrict__ in,
                                _Float16* __restrict__ out, int n) {
  int stride = gridDim.x * blockDim.x;
  for (int i = blockIdx.x * blockDim.x + threadIdx.x; i < n; i += stride)
    out[i] = (_Float16)in[i];
}

// in[R][C] f32 -> out[C][R] f16 (weights to [N][K] so B-fragments are linear)
__global__ void transpose_convert(const float* __restrict__ in,
                                  _Float16* __restrict__ out, int R, int C) {
  int n = R * C;
  int stride = gridDim.x * blockDim.x;
  for (int i = blockIdx.x * blockDim.x + threadIdx.x; i < n; i += stride) {
    int r = i / C, c = i - r * C;
    out[c * R + r] = (_Float16)in[i];
  }
}

// ---------------------------------------------------------------------------
// WMMA GEMM: C[M][N] = A[M][K](f16) * Bt[N][K](f16, pre-transposed) + bias[N]
// Block tile 128x128x32, 256 threads = 8 waves; wave tile 32x64 (2x4 WMMA).
// Double-buffered LDS; tiles staged by the Tensor Data Mover when available.
// ---------------------------------------------------------------------------
template <bool F16OUT>
__global__ __launch_bounds__(256)
void gemm_wmma(const _Float16* __restrict__ A, const _Float16* __restrict__ Bt,
               const float* __restrict__ bias, void* __restrict__ Cout,
               int M, int N, int K) {
  __shared__ __align__(16) _Float16 As[2][128 * 32];
  __shared__ __align__(16) _Float16 Bs[2][128 * 32];

  const int tid  = threadIdx.x;
  const int lane = tid & 31;
  const int wave = tid >> 5;
  const int wm   = wave & 3;   // 4 waves along M (32 rows each)
  const int wn   = wave >> 2;  // 2 waves along N (64 cols each)
  const int m0   = blockIdx.y * 128;
  const int n0   = blockIdx.x * 128;
  const int nk   = K >> 5;

  v8f acc[2][4] = {};

  // ---- Prologue: stage tile 0 into buffer 0 -------------------------------
#if HAVE_TDM
  if (tid == 0) {
    tdm_load_2d(LDS_OFF(As[0]), &A[(size_t)m0 * K], 32, 128, (unsigned)K);
    tdm_load_2d(LDS_OFF(Bs[0]), &Bt[(size_t)n0 * K], 32, 128, (unsigned)K);
  }
  __builtin_amdgcn_s_wait_tensorcnt(0);
#else
#pragma unroll
  for (int i = 0; i < 2; ++i) {
    int c = tid + i * 256;
    int row = c >> 2, col8 = (c & 3) * 8;
    *(v8h*)&As[0][row * 32 + col8] = *(const v8h*)&A[(size_t)(m0 + row) * K + col8];
    *(v8h*)&Bs[0][row * 32 + col8] = *(const v8h*)&Bt[(size_t)(n0 + row) * K + col8];
  }
#endif
  __syncthreads();

  int cur = 0;
  for (int t = 0; t < nk; ++t) {
    const int k0n  = (t + 1) << 5;
    const bool have = (t + 1) < nk;

    // ---- Kick off next tile into the alternate buffer ---------------------
#if HAVE_TDM
    if (have && tid == 0) {
      tdm_load_2d(LDS_OFF(As[cur ^ 1]), &A[(size_t)m0 * K + k0n], 32, 128, (unsigned)K);
      tdm_load_2d(LDS_OFF(Bs[cur ^ 1]), &Bt[(size_t)n0 * K + k0n], 32, 128, (unsigned)K);
    }
#else
    v8h ra[2], rb[2];
    if (have) {
#pragma unroll
      for (int i = 0; i < 2; ++i) {
        int c = tid + i * 256;
        int row = c >> 2, col8 = (c & 3) * 8;
        ra[i] = *(const v8h*)&A[(size_t)(m0 + row) * K + k0n + col8];
        rb[i] = *(const v8h*)&Bt[(size_t)(n0 + row) * K + k0n + col8];
      }
    }
#endif

    // ---- Compute on current buffer ----------------------------------------
    v16h a[2];
#pragma unroll
    for (int tm = 0; tm < 2; ++tm)
      a[tm] = load_a_frag(&As[cur][(wm * 32 + tm * 16) * 32], 32, lane);
#pragma unroll
    for (int tn = 0; tn < 4; ++tn) {
      v16h b = load_b_frag(&Bs[cur][(wn * 64 + tn * 16) * 32], 32, lane);
#pragma unroll
      for (int tm = 0; tm < 2; ++tm)
        acc[tm][tn] = __builtin_amdgcn_wmma_f32_16x16x32_f16(
            false, a[tm], false, b, (short)0, acc[tm][tn], false, false);
    }

    // ---- Publish next buffer ----------------------------------------------
#if HAVE_TDM
    __builtin_amdgcn_s_wait_tensorcnt(0);
#else
    if (have) {
#pragma unroll
      for (int i = 0; i < 2; ++i) {
        int c = tid + i * 256;
        int row = c >> 2, col8 = (c & 3) * 8;
        *(v8h*)&As[cur ^ 1][row * 32 + col8] = ra[i];
        *(v8h*)&Bs[cur ^ 1][row * 32 + col8] = rb[i];
      }
    }
#endif
    __syncthreads();
    cur ^= 1;
  }

  const int nco = lane & 15;
  const int hi  = lane >> 4;
#pragma unroll
  for (int tm = 0; tm < 2; ++tm) {
#pragma unroll
    for (int tn = 0; tn < 4; ++tn) {
      const int col = n0 + wn * 64 + tn * 16 + nco;
      const float bv = bias ? bias[col] : 0.0f;
#pragma unroll
      for (int r = 0; r < 8; ++r) {
        const int row = m0 + wm * 32 + tm * 16 + hi * 8 + r;
        const float v = acc[tm][tn][r] + bv;
        if (F16OUT)
          ((_Float16*)Cout)[(size_t)row * N + col] = (_Float16)v;
        else
          ((float*)Cout)[(size_t)row * N + col] = v;
      }
    }
  }
}

// ---------------------------------------------------------------------------
// Flash attention over qkv (f16, [B][L][3D]); out f16 [B][L][D].
// 128 threads = 4 waves; q-tile 64 (16 rows/wave); key-tile 64. LDS 56KB.
// Q/K tiles staged by TDM when available; V transposed manually into LDS.
// ---------------------------------------------------------------------------
__global__ __launch_bounds__(128)
void attn_kernel(const _Float16* __restrict__ qkv, _Float16* __restrict__ out) {
  __shared__ __align__(16) _Float16 Qs[64 * 128];    // [q][d]
  __shared__ __align__(16) _Float16 Ks[64 * 128];    // [key][d]
  __shared__ __align__(16) _Float16 Vts[128 * 64];   // [d][key] (transposed)
  __shared__ __align__(16) _Float16 Ps[4 * 16 * 64]; // per-wave P [q][key]

  const int tid  = threadIdx.x;
  const int lane = tid & 31;
  const int wave = tid >> 5;
  const int hi   = lane >> 4;
  const int nco  = lane & 15;
  const int qt = blockIdx.x, h = blockIdx.y, b = blockIdx.z;

  const size_t rs = 3 * (size_t)Dsz;  // qkv row stride in halves
  const _Float16* qbase = qkv + ((size_t)b * Lsz + qt * 64) * rs + h * HDsz;
  const _Float16* kbase = qkv + (size_t)b * Lsz * rs + Dsz + h * HDsz;
  const _Float16* vbase = kbase + Dsz;

  // Q tile: 64 rows x 128 halves.
#if HAVE_TDM
  if (tid == 0) tdm_load_2d(LDS_OFF(Qs), qbase, 128, 64, rs);
#else
#pragma unroll
  for (int i = 0; i < 8; ++i) {
    int c = tid + i * 128;
    int row = c >> 4, col8 = (c & 15) * 8;
    *(v8h*)&Qs[row * 128 + col8] = *(const v8h*)&qbase[row * rs + col8];
  }
#endif

  v8f acc[8] = {};   // O accumulator: [16q][128d] per wave, 8 d-tiles
  float mi[8], li[8];
#pragma unroll
  for (int r = 0; r < 8; ++r) { mi[r] = -1e30f; li[r] = 0.0f; }

  for (int kt = 0; kt < Lsz; kt += 64) {
    __syncthreads();  // all waves done with previous Ks/Vts
#if HAVE_TDM
    if (tid == 0)
      tdm_load_2d(LDS_OFF(Ks), kbase + (size_t)kt * rs, 128, 64, rs);
#else
#pragma unroll
    for (int i = 0; i < 8; ++i) {
      int c = tid + i * 128;
      int row = c >> 4, col8 = (c & 15) * 8;
      *(v8h*)&Ks[row * 128 + col8] =
          *(const v8h*)&kbase[(size_t)(kt + row) * rs + col8];
    }
#endif
    // V tile transposed into Vts[d][key].
#pragma unroll
    for (int i = 0; i < 8; ++i) {
      int c = tid + i * 128;
      int row = c >> 4, col8 = (c & 15) * 8;
      v8h v = *(const v8h*)&vbase[(size_t)(kt + row) * rs + col8];
#pragma unroll
      for (int j = 0; j < 8; ++j) Vts[(col8 + j) * 64 + row] = v[j];
    }
#if HAVE_TDM
    __builtin_amdgcn_s_wait_tensorcnt(0);  // Q (iter 0) + K DMA complete
#endif
    __syncthreads();

    // S = Q * K^T  (contraction over d=128 -> 4 WMMA k-steps per key subtile)
    v16h aq[4];
#pragma unroll
    for (int dk = 0; dk < 4; ++dk)
      aq[dk] = load_a_frag(&Qs[(wave * 16) * 128 + dk * 32], 128, lane);
    v8f s[4];
#pragma unroll
    for (int st = 0; st < 4; ++st) {
      s[st] = (v8f){};
#pragma unroll
      for (int dk = 0; dk < 4; ++dk) {
        v16h bk = load_b_frag(&Ks[(st * 16) * 128 + dk * 32], 128, lane);
        s[st] = __builtin_amdgcn_wmma_f32_16x16x32_f16(
            false, aq[dk], false, bk, (short)0, s[st], false, false);
      }
    }

    // Online softmax update per query row (row = hi*8 + r).
#pragma unroll
    for (int r = 0; r < 8; ++r) {
      float v0 = s[0][r] * SCALE, v1 = s[1][r] * SCALE;
      float v2 = s[2][r] * SCALE, v3 = s[3][r] * SCALE;
      float m = fmaxf(fmaxf(v0, v1), fmaxf(v2, v3));
      m = fmaxf(m, __shfl_xor(m, 1, 32));
      m = fmaxf(m, __shfl_xor(m, 2, 32));
      m = fmaxf(m, __shfl_xor(m, 4, 32));
      m = fmaxf(m, __shfl_xor(m, 8, 32));
      float mnew  = fmaxf(mi[r], m);
      float alpha = __expf(mi[r] - mnew);
      float p0 = __expf(v0 - mnew), p1 = __expf(v1 - mnew);
      float p2 = __expf(v2 - mnew), p3 = __expf(v3 - mnew);
      float sum = p0 + p1 + p2 + p3;
      sum += __shfl_xor(sum, 1, 32);
      sum += __shfl_xor(sum, 2, 32);
      sum += __shfl_xor(sum, 4, 32);
      sum += __shfl_xor(sum, 8, 32);
      li[r] = li[r] * alpha + sum;
      mi[r] = mnew;
#pragma unroll
      for (int dt = 0; dt < 8; ++dt) acc[dt][r] *= alpha;
      // Spill P to wave-private LDS (f16) for re-load as A fragments.
      int pofs = wave * 1024 + ((hi << 3) + r) * 64 + nco;
      Ps[pofs + 0]  = (_Float16)p0;
      Ps[pofs + 16] = (_Float16)p1;
      Ps[pofs + 32] = (_Float16)p2;
      Ps[pofs + 48] = (_Float16)p3;
    }

    // O += P * V  (contraction over 64 keys -> 2 WMMA k-steps)
#pragma unroll
    for (int ks = 0; ks < 2; ++ks) {
      v16h ap = load_a_frag(&Ps[wave * 1024 + ks * 32], 64, lane);
#pragma unroll
      for (int dt = 0; dt < 8; ++dt) {
        v16h bv = load_b_frag(&Vts[(dt * 16) * 64 + ks * 32], 64, lane);
        acc[dt] = __builtin_amdgcn_wmma_f32_16x16x32_f16(
            false, ap, false, bv, (short)0, acc[dt], false, false);
      }
    }
  }

  // Normalize and store O (f16) to [B][L][H*HD].
#pragma unroll
  for (int r = 0; r < 8; ++r) {
    const float inv = 1.0f / li[r];
    const size_t row = (size_t)b * Lsz + qt * 64 + wave * 16 + hi * 8 + r;
#pragma unroll
    for (int dt = 0; dt < 8; ++dt)
      out[row * Dsz + h * HDsz + dt * 16 + nco] = (_Float16)(acc[dt][r] * inv);
  }
}

// ---------------------------------------------------------------------------
// Host-side orchestration.
// ---------------------------------------------------------------------------
extern "C" void kernel_launch(void* const* d_in, const int* in_sizes, int n_in,
                              void* d_out, int out_size, void* d_ws, size_t ws_size,
                              hipStream_t stream) {
  const float* x      = (const float*)d_in[0];  // [B,L,D]
  const float* w_qkv  = (const float*)d_in[1];  // [D,3D]
  const float* b_qkv  = (const float*)d_in[2];  // [3D]
  const float* w_proj = (const float*)d_in[3];  // [D,D]
  const float* b_proj = (const float*)d_in[4];  // [D]

  char* ws = (char*)d_ws;
  _Float16* xh     = (_Float16*)(ws + 0);          //  32 MB  [8192][2048]
  _Float16* wqkvT  = (_Float16*)(ws + 33554432);   //  24 MB  [6144][2048]
  _Float16* wprojT = (_Float16*)(ws + 58720256);   //   8 MB  [2048][2048]
  _Float16* qkvh   = (_Float16*)(ws + 67108864);   //  96 MB  [8192][6144]
  _Float16* attnh  = (_Float16*)(ws + 167772160);  //  32 MB  [8192][2048]

  const int M = Bsz * Lsz;  // 8192

  convert_f32_f16<<<4096, 256, 0, stream>>>(x, xh, M * Dsz);
  transpose_convert<<<4096, 256, 0, stream>>>(w_qkv, wqkvT, Dsz, 3 * Dsz);
  transpose_convert<<<2048, 256, 0, stream>>>(w_proj, wprojT, Dsz, Dsz);

  dim3 g1(3 * Dsz / 128, M / 128);  // 48 x 64
  gemm_wmma<true><<<g1, 256, 0, stream>>>(xh, wqkvT, b_qkv, qkvh,
                                          M, 3 * Dsz, Dsz);

  dim3 g2(Lsz / 64, Hsz, Bsz);      // 32 x 16 x 4
  attn_kernel<<<g2, 128, 0, stream>>>(qkvh, attnh);

  dim3 g3(Dsz / 128, M / 128);      // 16 x 64
  gemm_wmma<false><<<g3, 256, 0, stream>>>(attnh, wprojT, b_proj, d_out,
                                           M, Dsz, Dsz);
}
